// AttentionGuidedIterativeBlock_76579266888376
// MI455X (gfx1250) — matrix operand order repaired and checked by reading, not compiled, not run on hardware
//
#include <hip/hip_runtime.h>
#include <hip/hip_bf16.h>
#include <math.h>
#include <stdint.h>

// ---------------- problem constants ----------------
#define D_   256
#define P_   32
#define I_   3
#define H_   8
#define B_   2
#define L_   2048
#define NT_  (B_ * L_)        // 4096 tokens
#define NC_  32               // chunks per batch along L
#define CL_  (L_ / NC_)       // 64 steps per chunk
#define CW_  (D_ + H_)        // 264 combined width
#define CWP_ 288              // 264 padded to multiple of 32 (K padding for w1)
#define PI_F 3.14159265358979323846f
#define INV_SQRT_P 0.17677669529663687f   // 1/sqrt(32)

typedef __attribute__((ext_vector_type(16))) _Float16 v16h;
typedef __attribute__((ext_vector_type(8)))  float    v8f;

// =====================================================================
// Weight prepack: f32 (K x N) row-major -> f16 WMMA-B fragment order:
//   packed[((kb*Ntiles + nt)*32 + lane)*16 + e], where the B 32x16 slice
//   element for lane is K = kb*32 + (lane>>4)*16 + e, N = nt*16 + (lane&15).
// Out-of-range (K,N) zero-filled.  Two b128 loads per lane per fragment.
// =====================================================================
__global__ void pack_w_kernel(const float* __restrict__ W, _Float16* __restrict__ out,
                              int K, int N, int Kpad, int Npad) {
  const int idx = blockIdx.x * blockDim.x + threadIdx.x;
  if (idx >= Kpad * Npad) return;
  const int e    = idx & 15;
  const int lane = (idx >> 4) & 31;
  const int tile = idx >> 9;          // kb*Ntiles + nt
  const int Ntiles = Npad >> 4;
  const int kb = tile / Ntiles, nt = tile % Ntiles;
  const int k = kb * 32 + ((lane >> 4) << 4) + e;
  const int n = (nt << 4) + (lane & 15);
  out[idx] = (_Float16)((k < K && n < N) ? W[(size_t)k * N + n] : 0.0f);
}

// =====================================================================
// Generic fused GEMM: out[M x N] = epi(A[M x K] @ Wp + bias)
//  - Wp is prepacked f16 (Kpad = 32*Ktiles rows, Npad = N cols, N % 16 == 0)
//  - NW 16x16 output tiles per wave along N (Ntiles % NW == 0)
//  - EPI_GATE: second K segment uses A2 (Wp rows [32*Ktiles, 32*(Ktiles+K2tiles)))
// =====================================================================
enum { EPI_NONE = 0, EPI_TANHPI = 1, EPI_GELU = 2, EPI_ACC = 3, EPI_GATE = 4, EPI_ADDX = 5 };

template <int EPI, int NW>
__global__ __launch_bounds__(256) void gemm_kernel(
    const float* __restrict__ A, const float* __restrict__ A2,
    const _Float16* __restrict__ Wp, const float* __restrict__ bias,
    float* __restrict__ out, const float* __restrict__ aux, float* __restrict__ aux2,
    int M, int N, int Nvalid, int Ktiles, int K2tiles,
    int ldA, int ldA2, int ldOut, int ldAux) {
  const int Mt = M >> 4;
  const int Ntiles = N >> 4;
  const int Ngrp = Ntiles / NW;
  const int wave = (blockIdx.x << 3) + (threadIdx.x >> 5);
  if (wave >= Mt * Ngrp) return;               // uniform per wave
  const int mt = wave / Ngrp, ng = wave % Ngrp;
  const int m0 = mt << 4, nt0 = ng * NW;
  const int lane = threadIdx.x & 31;
  const int row = lane & 15, hi = lane >> 4;

  v8f acc[NW];
#pragma unroll
  for (int w = 0; w < NW; ++w) acc[w] = (v8f){0.f, 0.f, 0.f, 0.f, 0.f, 0.f, 0.f, 0.f};

  const float* Ab  = A + (size_t)m0 * ldA;
  const float* A2b = (EPI == EPI_GATE) ? (A2 + (size_t)m0 * ldA2) : nullptr;
  const int KT = Ktiles + K2tiles;

  for (int kb = 0; kb < KT; ++kb) {
    const float* Ap;
    int ldcur;
    if (EPI == EPI_GATE && kb >= Ktiles) {
      Ap = A2b + (size_t)(kb - Ktiles) * 32;
      ldcur = ldA2;
    } else {
      Ap = Ab + (size_t)kb * 32;
      ldcur = ldA;
    }
    // A fragment: 16 contiguous-f32 loads (compiler fuses to b128), cvt to f16
    v16h af;
#pragma unroll
    for (int e = 0; e < 8; ++e) {
      af[e]     = (_Float16)Ap[(size_t)row * ldcur + (hi << 4) + e];
      af[8 + e] = (_Float16)Ap[(size_t)row * ldcur + (hi << 4) + 8 + e];
    }
    const _Float16* wrow = Wp + (((size_t)kb * Ntiles + nt0) * 32 + lane) * 16;
    if (kb + 1 < KT)
      __builtin_prefetch(Wp + (((size_t)(kb + 1) * Ntiles + nt0) * 32 + lane) * 16, 0, 1);
#pragma unroll
    for (int w = 0; w < NW; ++w) {
      const v16h bf = *(const v16h*)(wrow + (size_t)w * (32 * 16));
      acc[w] = __builtin_amdgcn_wmma_f32_16x16x32_f16(false, af, false, bf, (short)0,
                                                      acc[w], false, false);
    }
  }

  // epilogue
#pragma unroll
  for (int w = 0; w < NW; ++w) {
    const int gcol = ((nt0 + w) << 4) + row;
    const bool cok = (gcol < Nvalid);
    const float bv = cok ? bias[gcol] : 0.0f;
#pragma unroll
    for (int r = 0; r < 8; ++r) {
      const int grow = m0 + r + (hi << 3);
      float val = acc[w][r] + bv;
      if (EPI == EPI_TANHPI) {
        val = tanhf(val) * PI_F;
      } else if (EPI == EPI_GELU) {
        val = 0.5f * val * (1.0f + erff(val * 0.70710678118654752f));
      } else if (EPI == EPI_ACC) {
        if (cok) aux2[(size_t)grow * ldOut + gcol] += val;   // accumulated += refined
      } else if (EPI == EPI_GATE) {
        if (cok) val = aux[(size_t)grow * ldAux + gcol] + tanhf(val);  // q + tanh(.)
      } else if (EPI == EPI_ADDX) {
        if (cok) val = aux[(size_t)grow * ldAux + gcol] + val;         // x + .
      }
      if (cok) out[(size_t)grow * ldOut + gcol] = val;
    }
  }
}

// ---------------- init: query0 = x, accum = 0 ----------------
__global__ void init_kernel(const float* __restrict__ x, float* __restrict__ q0,
                            float* __restrict__ accum) {
  const int i = blockIdx.x * blockDim.x + threadIdx.x;
  if (i < NT_ * D_) { q0[i] = x[i]; accum[i] = 0.0f; }
}

// ---------------- per-chunk partial sums of cos(ph)*v / sin(ph)*v ----------------
__global__ __launch_bounds__(256) void chunksum_kernel(
    const float* __restrict__ phases, const float* __restrict__ values,
    float* __restrict__ pref_re, float* __restrict__ pref_im) {
  __shared__ float bc[P_], bs[P_];
  const int blk = blockIdx.x;                   // b*NC_ + c
  const int b = blk / NC_, c = blk % NC_;
  const int tid = threadIdx.x;                  // = d
  float sre[P_], sim[P_];
#pragma unroll
  for (int p = 0; p < P_; ++p) { sre[p] = 0.f; sim[p] = 0.f; }
  for (int j = 0; j < CL_; ++j) {
    const int t = b * L_ + c * CL_ + j;
    if (tid < P_) {
      const float ph = phases[(size_t)t * P_ + tid];
      bc[tid] = cosf(ph);
      bs[tid] = sinf(ph);
    }
    __syncthreads();
    const float v = values[(size_t)t * D_ + tid];
#pragma unroll
    for (int p = 0; p < P_; ++p) { sre[p] += bc[p] * v; sim[p] += bs[p] * v; }
    __syncthreads();
  }
#pragma unroll
  for (int p = 0; p < P_; ++p) {
    pref_re[(size_t)blk * P_ * D_ + p * D_ + tid] = sre[p];
    pref_im[(size_t)blk * P_ * D_ + p * D_ + tid] = sim[p];
  }
}

// ---------------- exclusive scan over chunks (per b,p,d chain) ----------------
__global__ void scan_kernel(float* __restrict__ pref_re, float* __restrict__ pref_im) {
  const int idx = blockIdx.x * blockDim.x + threadIdx.x;   // b*P*D + pd
  if (idx >= B_ * P_ * D_) return;
  const int b = idx / (P_ * D_), pd = idx % (P_ * D_);
  float rr = 0.f, ri = 0.f;
  for (int c = 0; c < NC_; ++c) {
    const size_t o = ((size_t)(b * NC_ + c)) * P_ * D_ + pd;
    const float tr = pref_re[o]; pref_re[o] = rr; rr += tr;
    const float ti = pref_im[o]; pref_im[o] = ri; ri += ti;
  }
}

// ---------------- softmax over H=8 head logits (logits stride 16, padded) ------
__global__ void softmax_kernel(const float* __restrict__ logits, float* __restrict__ attn) {
  const int t = blockIdx.x * blockDim.x + threadIdx.x;
  if (t >= NT_) return;
  float m = -3.0e38f;
  float e[H_];
#pragma unroll
  for (int h = 0; h < H_; ++h) m = fmaxf(m, logits[(size_t)t * 16 + h]);
  float s = 0.f;
#pragma unroll
  for (int h = 0; h < H_; ++h) { e[h] = expf(logits[(size_t)t * 16 + h] - m); s += e[h]; }
  const float inv = 1.0f / s;
#pragma unroll
  for (int h = 0; h < H_; ++h) attn[(size_t)t * H_ + h] = e[h] * inv;
}

// ---------------- retrieval: replay chunk cumsum in LDS, dot with query phasors ----
// dynamic LDS: mem_re[P*D] | mem_im[P*D] | bc[P] | bs[P] | qc[P] | qs[P]  (66 KB)
__global__ __launch_bounds__(256) void retrieve_kernel(
    const float* __restrict__ phases, const float* __restrict__ values,
    const float* __restrict__ qph, const float* __restrict__ pref_re,
    const float* __restrict__ pref_im, float* __restrict__ retrieved) {
  extern __shared__ float smem[];
  float* mem_re = smem;
  float* mem_im = smem + P_ * D_;
  float* bc = mem_im + P_ * D_;
  float* bs = bc + P_;
  float* qc = bs + P_;
  float* qs = qc + P_;
  const int blk = blockIdx.x;                  // b*NC_ + c
  const int b = blk / NC_, c = blk % NC_;
  const int tid = threadIdx.x;                 // = d
  for (int i = tid; i < P_ * D_; i += 256) {
    mem_re[i] = pref_re[(size_t)blk * P_ * D_ + i];
    mem_im[i] = pref_im[(size_t)blk * P_ * D_ + i];
  }
  __syncthreads();
  for (int j = 0; j < CL_; ++j) {
    const int l = c * CL_ + j;
    const int t = b * L_ + l;
    if (tid < P_) {
      const float ph = phases[(size_t)t * P_ + tid];
      bc[tid] = cosf(ph);
      bs[tid] = sinf(ph);
    } else if (tid < 2 * P_) {
      const int p = tid - P_;
      const float qp = qph[(size_t)t * P_ + p];
      qc[p] = cosf(qp);
      qs[p] = sinf(qp);
    }
    __syncthreads();
    const float v = values[(size_t)t * D_ + tid];
    float r = 0.f;
#pragma unroll
    for (int p = 0; p < P_; ++p) {
      const float mr = mem_re[p * D_ + tid] + bc[p] * v;
      mem_re[p * D_ + tid] = mr;
      const float mi = mem_im[p * D_ + tid] + bs[p] * v;
      mem_im[p * D_ + tid] = mi;
      r += qc[p] * mr + qs[p] * mi;
    }
    retrieved[(size_t)t * D_ + tid] = r * rsqrtf((float)(l + 1)) * INV_SQRT_P;
    __syncthreads();
  }
}

// ---------------- layernorm over [src1(256) | src2(w2)], zero-pads to ldOut ----
__global__ __launch_bounds__(256) void ln_kernel(
    const float* __restrict__ src1, const float* __restrict__ src2, int w2,
    const float* __restrict__ g, const float* __restrict__ bta,
    float* __restrict__ out, int ldOut) {
  __shared__ float r1[256], r2[256];
  const int t = blockIdx.x;
  const int tid = threadIdx.x;
  const int W = 256 + w2;
  const float x1 = src1[(size_t)t * 256 + tid];
  const float x2 = (tid < w2) ? src2[(size_t)t * w2 + tid] : 0.0f;
  r1[tid] = x1 + x2;
  r2[tid] = x1 * x1 + x2 * x2;
  __syncthreads();
  for (int s = 128; s > 0; s >>= 1) {
    if (tid < s) { r1[tid] += r1[tid + s]; r2[tid] += r2[tid + s]; }
    __syncthreads();
  }
  const float mean = r1[0] / (float)W;
  const float var = r2[0] / (float)W - mean * mean;
  const float inv = rsqrtf(var + 1e-5f);
  out[(size_t)t * ldOut + tid] = (x1 - mean) * inv * g[tid] + bta[tid];
  if (tid < w2)
    out[(size_t)t * ldOut + 256 + tid] = (x2 - mean) * inv * g[256 + tid] + bta[256 + tid];
  const int e = 256 + w2 + tid;                 // zero K-padding columns
  if (e < ldOut) out[(size_t)t * ldOut + e] = 0.0f;
}

// ---------------- host side ----------------
static inline int gemm_blocks(int M, int Npad, int NW) {
  const int waves = (M / 16) * ((Npad / 16) / NW);
  return (waves + 7) / 8;
}
static inline int pack_blocks(int Kpad, int Npad) {
  return (Kpad * Npad + 255) / 256;
}

extern "C" void kernel_launch(void* const* d_in, const int* in_sizes, int n_in,
                              void* d_out, int out_size, void* d_ws, size_t ws_size,
                              hipStream_t stream) {
  (void)in_sizes; (void)n_in; (void)out_size; (void)ws_size;
  const float* x        = (const float*)d_in[0];
  const float* pe_w     = (const float*)d_in[1];
  const float* pe_b     = (const float*)d_in[2];
  const float* tv_w     = (const float*)d_in[3];
  const float* tv_b     = (const float*)d_in[4];
  const float* mq_w     = (const float*)d_in[5];
  const float* mq_b     = (const float*)d_in[6];
  const float* ref_ln_g = (const float*)d_in[7];
  const float* ref_ln_b = (const float*)d_in[8];
  const float* ref_w1   = (const float*)d_in[9];
  const float* ref_b1   = (const float*)d_in[10];
  const float* ref_w2   = (const float*)d_in[11];
  const float* ref_b2   = (const float*)d_in[12];
  const float* gate_w   = (const float*)d_in[13];
  const float* gate_b   = (const float*)d_in[14];
  const float* out_ln_g = (const float*)d_in[15];
  const float* out_ln_b = (const float*)d_in[16];
  const float* out_w    = (const float*)d_in[17];
  const float* out_b    = (const float*)d_in[18];
  float* out = (float*)d_out;

  // ---- f32 workspace ----
  float* w = (float*)d_ws;
  float* values    = w; w += (size_t)NT_ * D_;
  float* phases    = w; w += (size_t)NT_ * P_;
  float* qph       = w; w += (size_t)NT_ * P_;
  float* logits    = w; w += (size_t)NT_ * 16;       // padded stride 16
  float* attn      = w; w += (size_t)NT_ * H_;
  float* pref_re   = w; w += (size_t)B_ * NC_ * P_ * D_;
  float* pref_im   = w; w += (size_t)B_ * NC_ * P_ * D_;
  float* q0        = w; w += (size_t)NT_ * D_;
  float* q1        = w; w += (size_t)NT_ * D_;
  float* retrieved = w; w += (size_t)NT_ * D_;
  float* lnc       = w; w += (size_t)NT_ * CWP_;     // stride 288, zero-padded
  float* hbuf      = w; w += (size_t)NT_ * (2 * D_);
  float* refined   = w; w += (size_t)NT_ * D_;
  float* accum     = w; w += (size_t)NT_ * D_;
  float* lnacc     = w; w += (size_t)NT_ * D_;

  // ---- f16 prepacked weights (64B aligned) ----
  uintptr_t hb = ((uintptr_t)w + 63) & ~(uintptr_t)63;
  _Float16* hp = (_Float16*)hb;
  _Float16* tvp = hp; hp += (size_t)256 * 256;
  _Float16* pep = hp; hp += (size_t)256 * 32;
  _Float16* mqp = hp; hp += (size_t)256 * 16;
  _Float16* w1p = hp; hp += (size_t)I_ * CWP_ * 512;
  _Float16* w2p = hp; hp += (size_t)I_ * 512 * 256;
  _Float16* gwp = hp; hp += (size_t)I_ * 512 * 256;
  _Float16* owp = hp; hp += (size_t)256 * 256;

  // ---- prepack all weights ----
  pack_w_kernel<<<pack_blocks(256, 256), 256, 0, stream>>>(tv_w, tvp, 256, 256, 256, 256);
  pack_w_kernel<<<pack_blocks(256, 32), 256, 0, stream>>>(pe_w, pep, 256, 32, 256, 32);
  pack_w_kernel<<<pack_blocks(256, 16), 256, 0, stream>>>(mq_w, mqp, 256, 8, 256, 16);
  for (int i = 0; i < I_; ++i) {
    pack_w_kernel<<<pack_blocks(CWP_, 512), 256, 0, stream>>>(
        ref_w1 + (size_t)i * CW_ * 512, w1p + (size_t)i * CWP_ * 512, CW_, 512, CWP_, 512);
    pack_w_kernel<<<pack_blocks(512, 256), 256, 0, stream>>>(
        ref_w2 + (size_t)i * 512 * 256, w2p + (size_t)i * 512 * 256, 512, 256, 512, 256);
    pack_w_kernel<<<pack_blocks(512, 256), 256, 0, stream>>>(
        gate_w + (size_t)i * 512 * 256, gwp + (size_t)i * 512 * 256, 512, 256, 512, 256);
  }
  pack_w_kernel<<<pack_blocks(256, 256), 256, 0, stream>>>(out_w, owp, 256, 256, 256, 256);

  // values = x @ tv_w + tv_b
  gemm_kernel<EPI_NONE, 4><<<gemm_blocks(NT_, 256, 4), 256, 0, stream>>>(
      x, nullptr, tvp, tv_b, values, nullptr, nullptr,
      NT_, 256, 256, 8, 0, D_, 0, D_, 0);
  // phases = tanh(x @ pe_w + pe_b) * pi
  gemm_kernel<EPI_TANHPI, 2><<<gemm_blocks(NT_, 32, 2), 256, 0, stream>>>(
      x, nullptr, pep, pe_b, phases, nullptr, nullptr,
      NT_, 32, 32, 8, 0, D_, 0, P_, 0);
  // chunk partial sums + exclusive scan over chunks
  chunksum_kernel<<<B_ * NC_, 256, 0, stream>>>(phases, values, pref_re, pref_im);
  scan_kernel<<<(B_ * P_ * D_ + 255) / 256, 256, 0, stream>>>(pref_re, pref_im);
  // query0 = x, accum = 0
  init_kernel<<<(NT_ * D_ + 255) / 256, 256, 0, stream>>>(x, q0, accum);

  const size_t ret_lds = (size_t)(2 * P_ * D_ + 4 * P_) * sizeof(float);  // 66 KB
  float* qc = q0;
  float* qn = q1;
  for (int i = 0; i < I_; ++i) {
    // q_phases = tanh(query @ pe_w + pe_b) * pi
    gemm_kernel<EPI_TANHPI, 2><<<gemm_blocks(NT_, 32, 2), 256, 0, stream>>>(
        qc, nullptr, pep, pe_b, qph, nullptr, nullptr,
        NT_, 32, 32, 8, 0, D_, 0, P_, 0);
    // logits = query @ mq_w + mq_b ; attn = softmax(logits)
    gemm_kernel<EPI_NONE, 1><<<gemm_blocks(NT_, 16, 1), 256, 0, stream>>>(
        qc, nullptr, mqp, mq_b, logits, nullptr, nullptr,
        NT_, 16, H_, 8, 0, D_, 0, 16, 0);
    softmax_kernel<<<(NT_ + 255) / 256, 256, 0, stream>>>(logits, attn);
    // retrieved (cumsum replay in LDS)
    retrieve_kernel<<<B_ * NC_, 256, ret_lds, stream>>>(
        phases, values, qph, pref_re, pref_im, retrieved);
    // lnc = LN([retrieved | attn]) with zero K-padding to 288
    ln_kernel<<<NT_, 256, 0, stream>>>(retrieved, attn, H_,
                                       ref_ln_g + (size_t)i * CW_,
                                       ref_ln_b + (size_t)i * CW_, lnc, CWP_);
    // h = gelu(lnc @ w1 + b1)   (Kpad = 288 -> 9 k-tiles)
    gemm_kernel<EPI_GELU, 4><<<gemm_blocks(NT_, 512, 4), 256, 0, stream>>>(
        lnc, nullptr, w1p + (size_t)i * CWP_ * 512, ref_b1 + (size_t)i * 512,
        hbuf, nullptr, nullptr, NT_, 512, 512, 9, 0, CWP_, 0, 512, 0);
    // refined = h @ w2 + b2 ; accum += refined
    gemm_kernel<EPI_ACC, 4><<<gemm_blocks(NT_, 256, 4), 256, 0, stream>>>(
        hbuf, nullptr, w2p + (size_t)i * 512 * 256, ref_b2 + (size_t)i * D_,
        refined, nullptr, accum, NT_, 256, 256, 16, 0, 2 * D_, 0, D_, 0);
    // query' = query + tanh([query|refined] @ gate_w + gate_b)
    gemm_kernel<EPI_GATE, 4><<<gemm_blocks(NT_, 256, 4), 256, 0, stream>>>(
        qc, refined, gwp + (size_t)i * 512 * 256, gate_b + (size_t)i * D_,
        qn, qc, nullptr, NT_, 256, 256, 8, 8, D_, D_, D_, D_);
    float* tmp = qc; qc = qn; qn = tmp;
  }
  // out = x + LN(accum) @ out_w + out_b
  ln_kernel<<<NT_, 256, 0, stream>>>(accum, nullptr, 0, out_ln_g, out_ln_b, lnacc, D_);
  gemm_kernel<EPI_ADDX, 4><<<gemm_blocks(NT_, 256, 4), 256, 0, stream>>>(
      lnacc, nullptr, owp, out_b, out, x, nullptr,
      NT_, 256, 256, 8, 0, D_, 0, D_, D_);
}
